// CannyLoss_28243704938553
// MI455X (gfx1250) — compile-verified, matrix operand order checked by reading, not run on hardware
//
#include <hip/hip_runtime.h>
#include <hip/hip_bf16.h>

// Problem constants (match reference)
#define BB 8
#define CC 19
#define HH 512
#define WW 512
#define HW (HH * WW)           // 262144 = 2^18
#define NIMG (2 * BB)          // pred-edges batch [0..7], label-edges batch [8..15]

typedef _Float16 v16h __attribute__((ext_vector_type(16)));
typedef float    v8f  __attribute__((ext_vector_type(8)));

// ---------------------------------------------------------------------------
// Kernel 0: zero the output scalar (harness poisons d_out with 0xAA)
// ---------------------------------------------------------------------------
__global__ void zero_out_kernel(float* out, int n) {
    int i = blockIdx.x * blockDim.x + threadIdx.x;
    if (i < n) out[i] = 0.0f;
}

// ---------------------------------------------------------------------------
// Kernel 1: per-pixel argmax over 19 channels -> u8 image; floor(labels*255)
// img layout: [2B, H, W] u8. batches 0..7 = pred image, 8..15 = label image
// ---------------------------------------------------------------------------
__global__ void prep_kernel(const float* __restrict__ pred,
                            const float* __restrict__ labels,
                            unsigned char* __restrict__ img) {
    int p = blockIdx.x * blockDim.x + threadIdx.x;
    if (p >= BB * HW) return;
    int b   = p >> 18;          // / HW
    int rem = p & (HW - 1);
    const float* pb = pred + (size_t)b * CC * HW + rem;
    float best = pb[0];
    int   cls  = 0;
#pragma unroll 4
    for (int c = 1; c < CC; ++c) {
        if (c + 1 < CC) __builtin_prefetch(pb + (size_t)(c + 1) * HW, 0, 1);
        float v = pb[(size_t)c * HW];
        if (v > best) { best = v; cls = c; }   // strict > keeps first max (jnp.argmax)
    }
    img[p] = (unsigned char)((cls * 255) & 255);            // (cls*255) % 256
    float lv = labels[p];
    img[(size_t)BB * HW + p] = (unsigned char)floorf(lv * 255.0f);
}

// ---------------------------------------------------------------------------
// Kernel 2: Sobel + L1 magnitude + quantized-direction NMS + double threshold.
// 16x16 output tile; image staged in LDS with halo 2 (edge-replicate pad),
// magnitude computed on 18x18 (zero outside the image, matching mp zero-pad).
// Emits weak, strong, and s0 (= strong) maps.
// ---------------------------------------------------------------------------
__global__ void sobel_nms_kernel(const unsigned char* __restrict__ img,
                                 unsigned char* __restrict__ weak,
                                 unsigned char* __restrict__ strg,
                                 unsigned char* __restrict__ s0) {
    __shared__ float t[20][20];
    __shared__ float m[18][18];
    const int bz = blockIdx.z;
    const unsigned char* im = img + (size_t)bz * HW;
    const int x0 = blockIdx.x * 16 - 2;
    const int y0 = blockIdx.y * 16 - 2;
    const int tx = threadIdx.x, ty = threadIdx.y;
    const int tid = ty * 16 + tx;

    // load 20x20 image tile, edge-clamped (mode='edge' padding)
    for (int i = tid; i < 400; i += 256) {
        int r = i / 20, c = i % 20;
        int y = y0 + r, x = x0 + c;
        int cy = y < 0 ? 0 : (y >= HH ? HH - 1 : y);
        int cx = x < 0 ? 0 : (x >= WW ? WW - 1 : x);
        t[r][c] = (float)im[(size_t)cy * WW + cx];
    }
    __syncthreads();

    // magnitude on 18x18 (center + 1 halo); zero outside image (zero-pad of mp)
    for (int i = tid; i < 324; i += 256) {
        int r = i / 18, c = i % 18;
        int y = y0 + 1 + r, x = x0 + 1 + c;
        float mv = 0.0f;
        if (y >= 0 && y < HH && x >= 0 && x < WW) {
            float a00 = t[r][c],     a01 = t[r][c + 1],     a02 = t[r][c + 2];
            float a10 = t[r + 1][c],                        a12 = t[r + 1][c + 2];
            float a20 = t[r + 2][c], a21 = t[r + 2][c + 1], a22 = t[r + 2][c + 2];
            float gx = (a02 - a00) + 2.0f * (a12 - a10) + (a22 - a20);
            float gy = (a20 - a00) + 2.0f * (a21 - a01) + (a22 - a02);
            mv = fabsf(gx) + fabsf(gy);
        }
        m[r][c] = mv;
    }
    __syncthreads();

    // per-thread NMS at its center pixel
    const int y = blockIdx.y * 16 + ty;
    const int x = blockIdx.x * 16 + tx;
    // recompute gradient at center (t center = t[ty+2][tx+2])
    float a00 = t[ty + 1][tx + 1], a01 = t[ty + 1][tx + 2], a02 = t[ty + 1][tx + 3];
    float a10 = t[ty + 2][tx + 1],                          a12 = t[ty + 2][tx + 3];
    float a20 = t[ty + 3][tx + 1], a21 = t[ty + 3][tx + 2], a22 = t[ty + 3][tx + 3];
    float gx = (a02 - a00) + 2.0f * (a12 - a10) + (a22 - a20);
    float gy = (a20 - a00) + 2.0f * (a21 - a01) + (a22 - a02);
    float ax = fabsf(gx), ay = fabsf(gy);
    float mg = m[ty + 1][tx + 1];

    bool keep;
    if (ay < 0.41421356f * ax) {                 // d0: ang < 22.5 or >= 157.5
        keep = (mg >= m[ty + 1][tx + 2]) && (mg >= m[ty + 1][tx]);
    } else if (ay >= 2.41421356f * ax) {         // d90
        keep = (mg >= m[ty + 2][tx + 1]) && (mg >= m[ty][tx + 1]);
    } else if (gx * gy >= 0.0f) {                // d45
        keep = (mg >= m[ty][tx + 2]) && (mg >= m[ty + 2][tx]);
    } else {                                     // d135
        keep = (mg >= m[ty][tx]) && (mg >= m[ty + 2][tx + 2]);
    }
    unsigned char sv = (keep && (mg > 200.0f)) ? 1 : 0;
    unsigned char wv = (keep && (mg > 100.0f)) ? 1 : 0;
    size_t g = (size_t)bz * HW + (size_t)y * WW + x;
    weak[g] = wv;
    strg[g] = sv;
    s0[g]   = sv;   // hysteresis init: s = strong
}

// ---------------------------------------------------------------------------
// Kernel 3: hysteresis, 8 propagation steps per launch.
// 32x32 output tile + 8-pixel halo => 48x48 LDS working set.
// step: s' = (weak & dilate3x3(s)) | strong (dilate zero-padded, as reference)
// After 8 steps the central 32x32 equals 8 exact global steps.
// ---------------------------------------------------------------------------
#define TS   32
#define HALO 8
#define LT   (TS + 2 * HALO)   // 48
__global__ void hyst_pass_kernel(const unsigned char* __restrict__ sin,
                                 unsigned char* __restrict__ sout,
                                 const unsigned char* __restrict__ weak,
                                 const unsigned char* __restrict__ strg) {
    __shared__ unsigned char w_[LT * LT];
    __shared__ unsigned char st_[LT * LT];
    __shared__ unsigned char a_[LT * LT];
    __shared__ unsigned char b_[LT * LT];
    const int bz = blockIdx.z;
    const size_t base = (size_t)bz * HW;
    const int x0 = blockIdx.x * TS - HALO;
    const int y0 = blockIdx.y * TS - HALO;
    const int tid = threadIdx.x;

    for (int i = tid; i < LT * LT; i += 256) {
        int r = i / LT, c = i % LT;
        int y = y0 + r, x = x0 + c;
        unsigned char wv = 0, sv = 0, cv = 0;
        if (y >= 0 && y < HH && x >= 0 && x < WW) {
            size_t g = base + (size_t)y * WW + x;
            wv = weak[g]; sv = strg[g]; cv = sin[g];
        }
        w_[i] = wv; st_[i] = sv; a_[i] = cv;
    }
    __syncthreads();

    unsigned char* cur = a_;
    unsigned char* nxt = b_;
    for (int step = 0; step < HALO; ++step) {
        for (int i = tid; i < LT * LT; i += 256) {
            int r = i / LT, c = i % LT;
            int mx = 0;
#pragma unroll
            for (int dy = -1; dy <= 1; ++dy) {
#pragma unroll
                for (int dx = -1; dx <= 1; ++dx) {
                    int rr = r + dy, cc = c + dx;
                    if (rr >= 0 && rr < LT && cc >= 0 && cc < LT)
                        mx |= cur[rr * LT + cc];
                }
            }
            nxt[i] = (unsigned char)((w_[i] & (mx ? 1 : 0)) | st_[i]);
        }
        __syncthreads();
        unsigned char* tmp = cur; cur = nxt; nxt = tmp;
    }

    for (int i = tid; i < TS * TS; i += 256) {
        int r = i / TS, c = i % TS;
        int y = blockIdx.y * TS + r, x = blockIdx.x * TS + c;
        sout[base + (size_t)y * WW + x] = cur[(r + HALO) * LT + (c + HALO)];
    }
}

// ---------------------------------------------------------------------------
// Kernel 4: cross-entropy reduction via WMMA column sums.
// Per (b, w): n1p = sum_h ep, sel = sum_h el, scr = sum_h ep*el, then
//   loss += (sel * log(512 + n1p*(e-1)) - scr) / (B*W)
// Column sums: D = ones(16x32) x Btile(32x16) with v_wmma_f32_16x16x32_f16;
// every row of D holds the per-column sums; lanes 0..15 read D[0][n] (VGPR0).
// One wave per 16-wide w tile; 8 waves/block; B fragments staged via LDS.
// ---------------------------------------------------------------------------
__global__ void ce_wmma_kernel(const unsigned char* __restrict__ sfin,
                               float* __restrict__ out) {
    __shared__ __align__(16) unsigned char lep[8][32][16];
    __shared__ __align__(16) unsigned char lel[8][32][16];
    const int wave = threadIdx.x >> 5;
    const int lane = threadIdx.x & 31;
    const int b  = blockIdx.y;
    const int w0 = blockIdx.x * 128 + wave * 16;
    const unsigned char* epb = sfin + (size_t)b * HW;          // pred edges
    const unsigned char* elb = sfin + (size_t)(BB + b) * HW;   // label edges

    v16h ones;
#pragma unroll
    for (int j = 0; j < 16; ++j) ones[j] = (_Float16)1.0f;
    v8f cEp = {};
    v8f cEl = {};
    v8f cCr = {};

    for (int hc = 0; hc < 16; ++hc) {
        const int h = hc * 32 + lane;          // each lane loads one row (16B)
        *(uint4*)&lep[wave][lane][0] = *(const uint4*)(epb + (size_t)h * WW + w0);
        *(uint4*)&lel[wave][lane][0] = *(const uint4*)(elb + (size_t)h * WW + w0);
        __syncthreads();

        // B-fragment (32x16, f16): lane -> column n = lane&15, K base = lane&16,
        // v16h element j <-> K = kb + j (two K values per VGPR, lo/hi halves).
        const int n  = lane & 15;
        const int kb = lane & 16;
        v16h bep, bel, bcr;
#pragma unroll
        for (int j = 0; j < 16; ++j) {
            unsigned e = lep[wave][kb + j][n];
            unsigned l = lel[wave][kb + j][n];
            bep[j] = (_Float16)(float)e;
            bel[j] = (_Float16)(float)l;
            bcr[j] = (_Float16)(float)(e & l);
        }
        cEp = __builtin_amdgcn_wmma_f32_16x16x32_f16(false, ones, false, bep,
                                                     (short)0, cEp, false, false);
        cEl = __builtin_amdgcn_wmma_f32_16x16x32_f16(false, ones, false, bel,
                                                     (short)0, cEl, false, false);
        cCr = __builtin_amdgcn_wmma_f32_16x16x32_f16(false, ones, false, bcr,
                                                     (short)0, cCr, false, false);
        __syncthreads();
    }

    if (lane < 16) {
        float n1p = cEp[0];   // (M=0, N=lane): column sum for w = w0 + lane
        float sel = cEl[0];
        float scr = cCr[0];
        float logZ = logf(512.0f + n1p * 1.7182818284590452f);   // (e - 1)
        float contrib = (sel * logZ - scr) * (1.0f / (float)(BB * WW));
        atomicAdd(out, contrib);
    }
}

// ---------------------------------------------------------------------------
// Launch pipeline
// ---------------------------------------------------------------------------
extern "C" void kernel_launch(void* const* d_in, const int* in_sizes, int n_in,
                              void* d_out, int out_size, void* d_ws, size_t ws_size,
                              hipStream_t stream) {
    (void)in_sizes; (void)n_in; (void)ws_size;
    const float* pred   = (const float*)d_in[0];
    const float* labels = (const float*)d_in[1];
    float* out = (float*)d_out;

    unsigned char* ws   = (unsigned char*)d_ws;
    const size_t MAP = (size_t)NIMG * HW;       // 4 MiB per map set
    unsigned char* img  = ws + 0 * MAP;
    unsigned char* weak = ws + 1 * MAP;
    unsigned char* strg = ws + 2 * MAP;
    unsigned char* sA   = ws + 3 * MAP;
    unsigned char* sB   = ws + 4 * MAP;

    // 0) zero output scalar
    zero_out_kernel<<<dim3((out_size + 255) / 256), dim3(256), 0, stream>>>(out, out_size);

    // 1) argmax image + quantized labels (stacked batch of 16 u8 images)
    prep_kernel<<<dim3((BB * HW) / 256), dim3(256), 0, stream>>>(pred, labels, img);

    // 2) Sobel + NMS + thresholds -> weak/strong/s0
    sobel_nms_kernel<<<dim3(WW / 16, HH / 16, NIMG), dim3(16, 16), 0, stream>>>(
        img, weak, strg, sA);

    // 3) hysteresis: 32 ping-pong passes x 8 steps = 256 steps (reference cap)
    for (int p = 0; p < 32; ++p) {
        const unsigned char* sin = (p & 1) ? sB : sA;
        unsigned char*       so  = (p & 1) ? sA : sB;
        hyst_pass_kernel<<<dim3(WW / TS, HH / TS, NIMG), dim3(256), 0, stream>>>(
            sin, so, weak, strg);
    }
    // even number of passes -> final edge maps in sA

    // 4) cross-entropy reduction with WMMA column sums
    ce_wmma_kernel<<<dim3(WW / 128, BB), dim3(256), 0, stream>>>(sA, out);
}
